// PendulumHJBLoss_65214783422673
// MI455X (gfx1250) — matrix operand order repaired and checked by reading, not compiled, not run on hardware
//
#include <hip/hip_runtime.h>

// ---------------- constants ----------------
#define N_TOTAL   524288
#define HDIM      128
#define HPAD      136          // halfs per LDS row (272B, 16B aligned)
#define NW_MAIN   8            // waves per block (main kernel)
#define TILES     (N_TOTAL / 32)
#define EPSV      1.0e-4f

typedef _Float16 f16_t;
typedef __attribute__((ext_vector_type(16))) _Float16     v16h;
typedef __attribute__((ext_vector_type(2)))  _Float16     v2h;
typedef __attribute__((ext_vector_type(8)))  float        v8f;
typedef __attribute__((ext_vector_type(4)))  unsigned int u32x4;

union FragH { v16h h; u32x4 u[2]; };

// ---------------- LDS layout (byte offsets, all 16B aligned) ----------------
#define OFF_EW1   0          // enc W1  float[256]
#define OFF_EB1   1024       // enc b1  float[128]
#define OFF_EB2   1536       // enc b2  float[128]
#define OFF_EB3   2048       // enc b3  float[4]
#define OFF_DW1   2064
#define OFF_DB1   3088
#define OFF_DB2   3600
#define OFF_DB3   4112
#define OFF_EW2T  4128       // enc W2^T half[128][HPAD]
#define OFF_DW2T  38944
#define OFF_EW3T  73760      // enc W3^T (padded to 16 rows) half[16][HPAD]
#define OFF_DW3T  78112
#define OFF_WAVE  82464      // per-wave: h-buffer half[32][HPAD] + PQ float[64]
#define PER_WAVE  8960
#define SMEM_MAIN (OFF_WAVE + NW_MAIN * PER_WAVE)
#define SMEM_SYMP (OFF_WAVE + 4 * PER_WAVE)

// ---------------- helpers ----------------
__device__ __forceinline__ void lds_fence() {
  asm volatile("s_wait_dscnt 0" ::: "memory");
}

// CDNA5 has a hardware v_tanh_f32 (TRANS32, co-executes with XDL WMMA).
__device__ __forceinline__ float fast_tanh(float x) {
#if __has_builtin(__builtin_amdgcn_tanhf)
  return __builtin_amdgcn_tanhf(x);
#else
  // cheap fallback: tanh(x) = 1 - 2/(exp(2x)+1), rcp instead of IEEE divide
  x = fminf(10.0f, fmaxf(-10.0f, x));
  float e = __expf(2.0f * x);
  return fmaf(-2.0f, __builtin_amdgcn_rcpf(e + 1.0f), 1.0f);
#endif
}

__device__ __forceinline__ float waveSum(float v) {
  v += __shfl_xor(v, 16, 32);
  v += __shfl_xor(v, 8, 32);
  v += __shfl_xor(v, 4, 32);
  v += __shfl_xor(v, 2, 32);
  v += __shfl_xor(v, 1, 32);
  return v;
}

struct MLPParams {
  const float* W1;   // [2][128] row-major
  const float* b1;   // [128]
  const f16_t* w2t;  // [128][HPAD], w2t[n][k] = W2[k][n]
  const float* b2;   // [128]
  const f16_t* w3t;  // [16][HPAD], rows >=2 zero
  const float* b3;   // [2]
};

__device__ __forceinline__ v16h load_frag2(const f16_t* p0, const f16_t* p1) {
  FragH f;
  f.u[0] = *(const u32x4*)p0;
  f.u[1] = *(const u32x4*)p1;
  return f.h;
}

// One MLP (2 -> 128 -> 128 -> 2) over 32 samples held one-per-lane.
// hbuf: this wave's half[32][HPAD]; pqbuf: this wave's float[64].
__device__ void mlp32(float p, float q, const MLPParams prm,
                      f16_t* hbuf, float* pqbuf, int lane,
                      float& oP, float& oQ) {
  // ---- layer 1 (VALU, K=2): lane = sample row ----
  f16_t* myrow = hbuf + lane * HPAD;
#pragma unroll 4
  for (int j = 0; j < HDIM; j += 2) {
    float a0 = fast_tanh(fmaf(p, prm.W1[j],     fmaf(q, prm.W1[HDIM + j],     prm.b1[j])));
    float a1 = fast_tanh(fmaf(p, prm.W1[j + 1], fmaf(q, prm.W1[HDIM + j + 1], prm.b1[j + 1])));
    v2h hv;
    hv.x = (_Float16)a0; hv.y = (_Float16)a1;
    *(v2h*)(myrow + j) = hv;
  }
  lds_fence();

  const int col = lane & 15;     // N (and M) sub-index
  const int hs  = lane >> 4;     // lane half select

  // ---- load all layer-2 A fragments (2 M-tiles x 4 K-blocks) ----
  v16h A[2][4];
#pragma unroll
  for (int mt = 0; mt < 2; ++mt) {
#pragma unroll
    for (int kb = 0; kb < 4; ++kb) {
      const f16_t* r = hbuf + (mt * 16 + col) * HPAD + kb * 32 + hs * 8;
      A[mt][kb] = load_frag2(r, r + 16);
    }
  }

  // ---- layer 2: 16x16x32 WMMA, 8 N-tiles x 4 K-blocks x 2 M-tiles ----
#pragma unroll
  for (int nt = 0; nt < 8; ++nt) {
    float bias = prm.b2[nt * 16 + col];
    v8f c0, c1;
#pragma unroll
    for (int r = 0; r < 8; ++r) { c0[r] = bias; c1[r] = bias; }
#pragma unroll
    for (int kb = 0; kb < 4; ++kb) {
      const f16_t* br = prm.w2t + (nt * 16 + col) * HPAD + kb * 32 + hs * 8;
      v16h B = load_frag2(br, br + 16);
      c0 = __builtin_amdgcn_wmma_f32_16x16x32_f16(false, A[0][kb], false, B,
                                                  (short)0, c0, false, false);
      c1 = __builtin_amdgcn_wmma_f32_16x16x32_f16(false, A[1][kb], false, B,
                                                  (short)0, c1, false, false);
    }
    // tanh + write h2 back into hbuf (h1 fully consumed into A regs)
#pragma unroll
    for (int r = 0; r < 8; ++r) {
      int Mr = r + 8 * hs;
      hbuf[Mr * HPAD + nt * 16 + col]        = (f16_t)fast_tanh(c0[r]);
      hbuf[(16 + Mr) * HPAD + nt * 16 + col] = (f16_t)fast_tanh(c1[r]);
    }
  }
  lds_fence();

  // ---- layer 3: K=128, N padded to 16 (cols 0/1 real) ----
  float b3v = (col == 0) ? prm.b3[0] : ((col == 1) ? prm.b3[1] : 0.0f);
  v8f d0, d1;
#pragma unroll
  for (int r = 0; r < 8; ++r) { d0[r] = b3v; d1[r] = b3v; }
#pragma unroll
  for (int kb = 0; kb < 4; ++kb) {
    const f16_t* r0 = hbuf + col * HPAD + kb * 32 + hs * 8;
    const f16_t* r1 = hbuf + (16 + col) * HPAD + kb * 32 + hs * 8;
    const f16_t* rb = prm.w3t + col * HPAD + kb * 32 + hs * 8;
    v16h a0 = load_frag2(r0, r0 + 16);
    v16h a1 = load_frag2(r1, r1 + 16);
    v16h B  = load_frag2(rb, rb + 16);
    d0 = __builtin_amdgcn_wmma_f32_16x16x32_f16(false, a0, false, B,
                                                (short)0, d0, false, false);
    d1 = __builtin_amdgcn_wmma_f32_16x16x32_f16(false, a1, false, B,
                                                (short)0, d1, false, false);
  }
  // extract columns 0 (P) and 1 (Q) -> per-sample scalars via LDS
  if (col < 2) {
    float* dst = pqbuf + col * 32;
#pragma unroll
    for (int r = 0; r < 8; ++r) {
      dst[r + 8 * hs]      = d0[r];
      dst[16 + r + 8 * hs] = d1[r];
    }
  }
  lds_fence();
  oP = pqbuf[lane];
  oQ = pqbuf[32 + lane];
}

// ---------------- weight staging ----------------
__device__ void load_params(char* smem,
    const float* eW1, const float* eb1, const float* eW2, const float* eb2,
    const float* eW3, const float* eb3,
    const float* dW1, const float* db1, const float* dW2, const float* db2,
    const float* dW3, const float* db3) {
  const int tid = threadIdx.x, bs = blockDim.x;
  float* sEW1 = (float*)(smem + OFF_EW1);  float* sDW1 = (float*)(smem + OFF_DW1);
  float* sEB1 = (float*)(smem + OFF_EB1);  float* sDB1 = (float*)(smem + OFF_DB1);
  float* sEB2 = (float*)(smem + OFF_EB2);  float* sDB2 = (float*)(smem + OFF_DB2);
  float* sEB3 = (float*)(smem + OFF_EB3);  float* sDB3 = (float*)(smem + OFF_DB3);
  f16_t* sEW2T = (f16_t*)(smem + OFF_EW2T); f16_t* sDW2T = (f16_t*)(smem + OFF_DW2T);
  f16_t* sEW3T = (f16_t*)(smem + OFF_EW3T); f16_t* sDW3T = (f16_t*)(smem + OFF_DW3T);

  for (int i = tid; i < 2 * HDIM; i += bs) { sEW1[i] = eW1[i]; sDW1[i] = dW1[i]; }
  for (int i = tid; i < HDIM; i += bs) {
    sEB1[i] = eb1[i]; sEB2[i] = eb2[i]; sDB1[i] = db1[i]; sDB2[i] = db2[i];
  }
  if (tid < 2) { sEB3[tid] = eb3[tid]; sDB3[tid] = db3[tid]; }
  for (int i = tid; i < HDIM * HDIM; i += bs) {
    int k = i >> 7, n = i & 127;
    sEW2T[n * HPAD + k] = (f16_t)eW2[i];
    sDW2T[n * HPAD + k] = (f16_t)dW2[i];
  }
  for (int i = tid; i < 16 * HDIM; i += bs) {   // W3 is [128][2]; pad N to 16
    int n = i >> 7, k = i & 127;
    f16_t v = (f16_t)0.0f, w = (f16_t)0.0f;
    if (n < 2) { v = (f16_t)eW3[k * 2 + n]; w = (f16_t)dW3[k * 2 + n]; }
    sEW3T[n * HPAD + k] = v;
    sDW3T[n * HPAD + k] = w;
  }
}

// ---------------- kernels ----------------
__global__ void zero_ws(float* ws) {
  if (threadIdx.x < 8) ws[threadIdx.x] = 0.0f;
}

__global__ __launch_bounds__(256) void pendulum_main(
    const float* __restrict__ p0, const float* __restrict__ q0,
    const float* __restrict__ p1, const float* __restrict__ q1,
    const float* __restrict__ omega, const float* __restrict__ dt,
    const float* __restrict__ Ptrue, const float* __restrict__ Qtrue,
    const float* eW1, const float* eb1, const float* eW2, const float* eb2,
    const float* eW3, const float* eb3,
    const float* dW1, const float* db1, const float* dW2, const float* db2,
    const float* dW3, const float* db3,
    float* __restrict__ ws) {
  extern __shared__ char smem[];
  load_params(smem, eW1, eb1, eW2, eb2, eW3, eb3, dW1, db1, dW2, db2, dW3, db3);
  __syncthreads();

  const int lane = threadIdx.x & 31;
  const int wave = threadIdx.x >> 5;
  MLPParams enc = {(const float*)(smem + OFF_EW1), (const float*)(smem + OFF_EB1),
                   (const f16_t*)(smem + OFF_EW2T), (const float*)(smem + OFF_EB2),
                   (const f16_t*)(smem + OFF_EW3T), (const float*)(smem + OFF_EB3)};
  MLPParams dec = {(const float*)(smem + OFF_DW1), (const float*)(smem + OFF_DB1),
                   (const f16_t*)(smem + OFF_DW2T), (const float*)(smem + OFF_DB2),
                   (const f16_t*)(smem + OFF_DW3T), (const float*)(smem + OFF_DB3)};
  f16_t* hbuf  = (f16_t*)(smem + OFF_WAVE + wave * PER_WAVE);
  float* pqbuf = (float*)(smem + OFF_WAVE + wave * PER_WAVE + 32 * HPAD * 2);

  const float dtv = dt[0];
  const int wgid = blockIdx.x * NW_MAIN + wave;
  const int nw = gridDim.x * NW_MAIN;

  float s_recon = 0.f, s_cons = 0.f, s_evo = 0.f, s_g1 = 0.f, s_g2 = 0.f;

  for (int t = wgid; t < TILES; t += nw) {
    const int i = t * 32 + lane;
    float p0v = p0[i], q0v = q0[i];
    float P0, Q0; mlp32(p0v, q0v, enc, hbuf, pqbuf, lane, P0, Q0);
    float p1v = p1[i], q1v = q1[i];
    float P1, Q1; mlp32(p1v, q1v, enc, hbuf, pqbuf, lane, P1, Q1);
    float pr, qr; mlp32(P0, Q0, dec, hbuf, pqbuf, lane, pr, qr);

    float dp = p0v - pr, dq = q0v - qr;
    s_recon += dp * dp + dq * dq;
    float dc = P0 - P1;
    s_cons += dc * dc;
    s_evo += 1.0f - __cosf(Q1 - Q0 - omega[i] * dtv);
    float g = P0 - Ptrue[i];
    s_g1 += g * g;
    s_g2 += 1.0f - __cosf(Q0 - Qtrue[i]);
  }

  s_recon = waveSum(s_recon);
  s_cons  = waveSum(s_cons);
  s_evo   = waveSum(s_evo);
  s_g1    = waveSum(s_g1);
  s_g2    = waveSum(s_g2);
  if (lane == 0) {
    atomicAdd(ws + 0, s_recon);
    atomicAdd(ws + 1, s_cons);
    atomicAdd(ws + 2, s_evo);
    atomicAdd(ws + 3, s_g1);
    atomicAdd(ws + 4, s_g2);
  }
}

// 4 waves = 4 perturbation groups of the first 32 samples.
__global__ __launch_bounds__(128) void pendulum_symp(
    const float* __restrict__ p0, const float* __restrict__ q0,
    const float* eW1, const float* eb1, const float* eW2, const float* eb2,
    const float* eW3, const float* eb3,
    const float* dW1, const float* db1, const float* dW2, const float* db2,
    const float* dW3, const float* db3,
    float* __restrict__ ws) {
  extern __shared__ char smem[];
  load_params(smem, eW1, eb1, eW2, eb2, eW3, eb3, dW1, db1, dW2, db2, dW3, db3);
  __syncthreads();

  const int lane = threadIdx.x & 31;
  const int wave = threadIdx.x >> 5;  // group: 0:p+e 1:p-e 2:q+e 3:q-e
  MLPParams enc = {(const float*)(smem + OFF_EW1), (const float*)(smem + OFF_EB1),
                   (const f16_t*)(smem + OFF_EW2T), (const float*)(smem + OFF_EB2),
                   (const f16_t*)(smem + OFF_EW3T), (const float*)(smem + OFF_EB3)};
  f16_t* hbuf  = (f16_t*)(smem + OFF_WAVE + wave * PER_WAVE);
  float* pqbuf = (float*)(smem + OFF_WAVE + wave * PER_WAVE + 32 * HPAD * 2);

  float pi = p0[lane], qi = q0[lane];
  float pp = pi + ((wave == 0) ? EPSV : ((wave == 1) ? -EPSV : 0.0f));
  float qq = qi + ((wave == 2) ? EPSV : ((wave == 3) ? -EPSV : 0.0f));
  float Pg, Qg;
  mlp32(pp, qq, enc, hbuf, pqbuf, lane, Pg, Qg);
  // mlp32 left P at pqbuf[lane], Q at pqbuf[32+lane] for each wave's slice.
  __syncthreads();

  if (wave == 0) {
    const float* g0 = (const float*)(smem + OFF_WAVE + 0 * PER_WAVE + 32 * HPAD * 2);
    const float* g1 = (const float*)(smem + OFF_WAVE + 1 * PER_WAVE + 32 * HPAD * 2);
    const float* g2 = (const float*)(smem + OFF_WAVE + 2 * PER_WAVE + 32 * HPAD * 2);
    const float* g3 = (const float*)(smem + OFF_WAVE + 3 * PER_WAVE + 32 * HPAD * 2);
    const float inv2e = 1.0f / (2.0f * EPSV);
    float dPdp = (g0[lane] - g1[lane]) * inv2e;
    float dPdq = (g2[lane] - g3[lane]) * inv2e;
    float dQdp = (g0[32 + lane] - g1[32 + lane]) * inv2e;
    float dQdq = (g2[32 + lane] - g3[32 + lane]) * inv2e;
    float pb = dPdq * dQdp - dPdp * dQdq;
    float tsum = fabsf(pb) - 1.0f;
    tsum = waveSum(tsum * tsum);
    if (lane == 0) ws[5] = tsum;
  }
}

__global__ void pendulum_finalize(const float* __restrict__ ws,
                                  float* __restrict__ out) {
  if (threadIdx.x == 0) {
    const float invN = 1.0f / (float)N_TOTAL;
    float recon = ws[0] * invN;
    float cons  = ws[1] * invN;
    float evo   = ws[2] * invN;
    float gauge = (ws[3] + ws[4]) * invN;
    float symp  = ws[5] * (1.0f / 32.0f);
    float total = 1.0f * recon + 10.0f * cons + 5.0f * evo +
                  0.1f * symp + 5.0f * gauge;
    out[0] = total; out[1] = recon; out[2] = cons;
    out[3] = evo;   out[4] = symp;  out[5] = gauge;
  }
}

// ---------------- entry ----------------
extern "C" void kernel_launch(void* const* d_in, const int* in_sizes, int n_in,
                              void* d_out, int out_size, void* d_ws, size_t ws_size,
                              hipStream_t stream) {
  (void)in_sizes; (void)n_in; (void)out_size; (void)ws_size;
  const float* p0    = (const float*)d_in[0];
  const float* q0    = (const float*)d_in[1];
  const float* p1    = (const float*)d_in[2];
  const float* q1    = (const float*)d_in[3];
  const float* omega = (const float*)d_in[4];
  const float* dt    = (const float*)d_in[5];
  const float* Ptrue = (const float*)d_in[6];
  const float* Qtrue = (const float*)d_in[7];
  const float* eW1 = (const float*)d_in[8];
  const float* eb1 = (const float*)d_in[9];
  const float* eW2 = (const float*)d_in[10];
  const float* eb2 = (const float*)d_in[11];
  const float* eW3 = (const float*)d_in[12];
  const float* eb3 = (const float*)d_in[13];
  const float* dW1 = (const float*)d_in[14];
  const float* db1 = (const float*)d_in[15];
  const float* dW2 = (const float*)d_in[16];
  const float* db2 = (const float*)d_in[17];
  const float* dW3 = (const float*)d_in[18];
  const float* db3 = (const float*)d_in[19];
  float* ws  = (float*)d_ws;
  float* out = (float*)d_out;

  zero_ws<<<1, 32, 0, stream>>>(ws);
  pendulum_main<<<1024, 256, SMEM_MAIN, stream>>>(
      p0, q0, p1, q1, omega, dt, Ptrue, Qtrue,
      eW1, eb1, eW2, eb2, eW3, eb3, dW1, db1, dW2, db2, dW3, db3, ws);
  pendulum_symp<<<1, 128, SMEM_SYMP, stream>>>(
      p0, q0, eW1, eb1, eW2, eb2, eW3, eb3,
      dW1, db1, dW2, db2, dW3, db3, ws);
  pendulum_finalize<<<1, 32, 0, stream>>>(ws, out);
}
